// SelfAttention_77515569758295
// MI455X (gfx1250) — matrix operand order repaired and checked by reading, not compiled, not run on hardware
//
#include <hip/hip_runtime.h>
#include <hip/hip_bf16.h>

// Self-attention: B=2, S=4096, E=512, H=8, D=64. fp32 in/out, f16 WMMA inside.

typedef __attribute__((ext_vector_type(16))) _Float16 v16h;
typedef __attribute__((ext_vector_type(4)))  _Float16 v4h;
typedef __attribute__((ext_vector_type(8)))  float    v8f;

#define S_LEN 4096
#define EMB   512
#define NHEAD 8
#define HDIM  64
#define MTOT  8192   // B*S
#define LOG2E 1.44269504088896340736f

// ---------------------------------------------------------------------------
// Fragment loaders (wave32), this code's contiguous k order k = 16*(lane/16) + element, same for A and B
//
// A 16x32 f16 (MxK): lane L holds row M=L%16.
//   lanes 0-15 : halves 0-15 -> K=0..15
//   lanes 16-31: halves 0-15 -> K=16..31
__device__ inline v16h load_frag_a(const _Float16* base, int lda, int lane) {
  v16h a;
  const _Float16* row = base + (size_t)(lane & 15) * lda + 16 * (lane >> 4);
#pragma unroll
  for (int h = 0; h < 16; h += 2) {
    int k = h;
    a[h]     = row[k];
    a[h + 1] = row[k + 1];
  }
  return a;
}

// B 32x16 f16 (KxN): lane L holds column N=L%16.
//   lanes 0-15: K=0..15 ; lanes 16-31: K=16..31
// B tiles stored N-major in LDS -> 16 contiguous halves per lane (b128 loads).
__device__ inline v16h load_frag_b(const _Float16* baseNT, int ldb, int lane) {
  v16h b;
  const _Float16* row = baseNT + (size_t)(lane & 15) * ldb + 16 * (lane >> 4);
#pragma unroll
  for (int h = 0; h < 16; ++h) b[h] = row[h];
  return b;
}

#define WMMA_F16(A, B, C) \
  __builtin_amdgcn_wmma_f32_16x16x32_f16(false, (A), false, (B), (short)0, (C), false, false)

// xor-butterfly max over 16-lane groups via ds_swizzle_b32 (imm xor masks).
__device__ inline float swz_max16(float x) {
  x = fmaxf(x, __int_as_float(__builtin_amdgcn_ds_swizzle(__float_as_int(x), 0x041f)));
  x = fmaxf(x, __int_as_float(__builtin_amdgcn_ds_swizzle(__float_as_int(x), 0x081f)));
  x = fmaxf(x, __int_as_float(__builtin_amdgcn_ds_swizzle(__float_as_int(x), 0x101f)));
  x = fmaxf(x, __int_as_float(__builtin_amdgcn_ds_swizzle(__float_as_int(x), 0x201f)));
  return x;
}

// ---------------------------------------------------------------------------
__global__ __launch_bounds__(256) void cvt_f32_f16(const float* __restrict__ src,
                                                   _Float16* __restrict__ dst, int n4) {
  int i = blockIdx.x * 256 + threadIdx.x;
  if (i < n4) {
    float4 v = reinterpret_cast<const float4*>(src)[i];
    v4h h = {(_Float16)v.x, (_Float16)v.y, (_Float16)v.z, (_Float16)v.w};
    reinterpret_cast<v4h*>(dst)[i] = h;
  }
}

// ---------------------------------------------------------------------------
// GEMM: C[M,512] = A[M,512] (f16,[m][k]) x W^T, W given f16 row-major [n][k].
// Block tile 128x128, BK=32, 256 threads = 8 waves; wave w owns rows w*16..+16
// and all 128 columns (8 WMMA per k-step per wave). Double-buffered LDS with
// register-staged global prefetch; one barrier per k-step.
// MODE 0: f16 out reshaped to [B,H,S,D], scaled; coalesced via LDS C-staging.
// MODE 1: f32 out row-major; coalesced via LDS C-staging (two 64-row passes).
template <int MODE>
__global__ __launch_bounds__(256) void gemm_wmma(const _Float16* __restrict__ A,
                                                 const _Float16* __restrict__ W,
                                                 float scale,
                                                 _Float16* __restrict__ out16,
                                                 float* __restrict__ out32) {
  __shared__ _Float16 sA[2][128 * 32];  // [m][k]
  __shared__ _Float16 sB[2][128 * 32];  // [n][k]
  const int tid = threadIdx.x, lane = tid & 31, wave = tid >> 5;
  const int m0 = blockIdx.x * 128, n0 = blockIdx.y * 128;
  const int r = tid >> 1, c = (tid & 1) * 16;  // 2 threads/row, 16 halves each

  const float4* gA = reinterpret_cast<const float4*>(A + (size_t)(m0 + r) * EMB + c);
  const float4* gB = reinterpret_cast<const float4*>(W + (size_t)(n0 + r) * EMB + c);
  float4 a0 = gA[0], a1 = gA[1], b0 = gB[0], b1 = gB[1];

  v8f acc[8] = {};

  for (int k0 = 0, buf = 0; k0 < EMB; k0 += 32, buf ^= 1) {
    {  // commit staged registers to LDS
      float4* sa = reinterpret_cast<float4*>(&sA[buf][r * 32 + c]);
      sa[0] = a0;
      sa[1] = a1;
      float4* sb = reinterpret_cast<float4*>(&sB[buf][r * 32 + c]);
      sb[0] = b0;
      sb[1] = b1;
    }
    __syncthreads();
    if (k0 + 32 < EMB) {  // prefetch next tile while wmma runs
      gA += 4;  // 32 halves = 4 float4
      gB += 4;
      a0 = gA[0];
      a1 = gA[1];
      b0 = gB[0];
      b1 = gB[1];
    }
    v16h af = load_frag_a(&sA[buf][wave * 16 * 32], 32, lane);
#pragma unroll
    for (int j = 0; j < 8; ++j) {
      v16h bf = load_frag_b(&sB[buf][j * 16 * 32], 32, lane);
      acc[j] = WMMA_F16(af, bf, acc[j]);
    }
  }

  // C layout: VGPR i -> m_local = i + 8*(lane>>4), n_local = lane&15
  const int mh = 8 * (lane >> 4), nl = lane & 15;
  if constexpr (MODE == 0) {
    __shared__ _Float16 sC[128 * 128];  // C staging for coalesced stores
#pragma unroll
    for (int j = 0; j < 8; ++j)
#pragma unroll
      for (int i = 0; i < 8; ++i)
        sC[(wave * 16 + i + mh) * 128 + j * 16 + nl] = (_Float16)(acc[j][i] * scale);
    __syncthreads();
    // coalesced: thread covers row r, 64-col half (tid&1) = one head's d-range
    const int m = m0 + r;
    const int b = m >> 12, s = m & (S_LEN - 1);
    const int h = (n0 >> 6) + (tid & 1);
    const float4* src = reinterpret_cast<const float4*>(sC + r * 128 + (tid & 1) * 64);
    float4* dst = reinterpret_cast<float4*>(
        out16 + (((size_t)(b * NHEAD + h) * S_LEN) + s) * HDIM);
    dst[0] = src[0];
    dst[1] = src[1];
    dst[2] = src[2];
    dst[3] = src[3];
  } else {
    __shared__ float sCf[64 * 128];  // f32 C staging, two 64-row passes
#pragma unroll
    for (int half = 0; half < 2; ++half) {
      __syncthreads();
      if ((wave >> 2) == half) {
#pragma unroll
        for (int j = 0; j < 8; ++j)
#pragma unroll
          for (int i = 0; i < 8; ++i)
            sCf[((wave & 3) * 16 + i + mh) * 128 + j * 16 + nl] = acc[j][i] * scale;
      }
      __syncthreads();
      // coalesced: 4 threads/row, 32 f32 (8 float4) each
      const int rr = tid >> 2, cc = (tid & 3) * 32;
      const float4* src = reinterpret_cast<const float4*>(sCf + rr * 128 + cc);
      float4* dst =
          reinterpret_cast<float4*>(out32 + (size_t)(m0 + half * 64 + rr) * EMB + n0 + cc);
#pragma unroll
      for (int q = 0; q < 8; ++q) dst[q] = src[q];
    }
  }
}

// ---------------------------------------------------------------------------
// Flash attention. Q,K,V f16 [B*H, S, D]; Q pre-scaled by log2e/sqrt(D), so the
// whole softmax runs in the log2 domain (exp2f == raw v_exp_f32).
// Grid: (S/128, B*H). 256 threads = 8 waves; wave w owns 16 query rows.
// Double-buffered K/V tiles with register-staged prefetch. Row sums computed
// with the P @ ones WMMA trick (broadcast across columns = per-lane layout).
__global__ __launch_bounds__(256) void flash_attn(const _Float16* __restrict__ Q,
                                                  const _Float16* __restrict__ K,
                                                  const _Float16* __restrict__ V,
                                                  _Float16* __restrict__ Oout) {
  __shared__ _Float16 sK[2][64 * 64];   // [key][d]  (= B-frag layout for Q.K^T)
  __shared__ _Float16 sVt[2][64 * 64];  // [d][key]  (= B-frag layout for P.V)
  __shared__ _Float16 sP[8 * 16 * 64];  // per-wave P staging (C->A relayout)

  const int tid = threadIdx.x, lane = tid & 31, wave = tid >> 5;
  const int bh = blockIdx.y;
  const int q0 = blockIdx.x * 128 + wave * 16;

  const _Float16* Qb = Q + ((size_t)bh * S_LEN + q0) * HDIM;
  const v16h qa0 = load_frag_a(Qb, HDIM, lane);       // d = 0..31
  const v16h qa1 = load_frag_a(Qb + 32, HDIM, lane);  // d = 32..63

  v16h vones;
#pragma unroll
  for (int h = 0; h < 16; ++h) vones[h] = (_Float16)1.0f;

  v8f o[4] = {};
  float mrow[8], lrow[8], alphav[8];
#pragma unroll
  for (int i = 0; i < 8; ++i) { mrow[i] = -1e30f; lrow[i] = 0.0f; }

  _Float16* myP = sP + wave * (16 * 64);

  const int r = tid >> 2, c = (tid & 3) * 16;  // tile loaders: 4 thr/row
  const _Float16* Kbase = K + (size_t)bh * S_LEN * HDIM;
  const _Float16* Vbase = V + (size_t)bh * S_LEN * HDIM;

  float4 kr0, kr1;
  union { float4 f4[2]; _Float16 h[16]; } vr;
  {  // preload tile 0
    const float4* gk = reinterpret_cast<const float4*>(Kbase + (size_t)r * HDIM + c);
    kr0 = gk[0];
    kr1 = gk[1];
    const float4* gv = reinterpret_cast<const float4*>(Vbase + (size_t)r * HDIM + c);
    vr.f4[0] = gv[0];
    vr.f4[1] = gv[1];
  }

  for (int kt = 0, buf = 0; kt < S_LEN; kt += 64, buf ^= 1) {
    {  // commit staged K (natural) and V (transposed) to LDS
      float4* sk4 = reinterpret_cast<float4*>(&sK[buf][r * 64 + c]);
      sk4[0] = kr0;
      sk4[1] = kr1;
#pragma unroll
      for (int j = 0; j < 16; ++j) sVt[buf][(c + j) * 64 + r] = vr.h[j];
    }
    __syncthreads();
    if (kt + 64 < S_LEN) {  // prefetch next tile behind the wmma work
      const float4* gk =
          reinterpret_cast<const float4*>(Kbase + (size_t)(kt + 64 + r) * HDIM + c);
      kr0 = gk[0];
      kr1 = gk[1];
      const float4* gv =
          reinterpret_cast<const float4*>(Vbase + (size_t)(kt + 64 + r) * HDIM + c);
      vr.f4[0] = gv[0];
      vr.f4[1] = gv[1];
    }

    // S tile = Q x K^T : 16x64 per wave, f32 (already in log2 domain)
    v8f sf[4];
#pragma unroll
    for (int j = 0; j < 4; ++j) {
      v8f z = {};
      z = WMMA_F16(qa0, load_frag_b(&sK[buf][j * 16 * 64], 64, lane), z);
      z = WMMA_F16(qa1, load_frag_b(&sK[buf][j * 16 * 64 + 32], 64, lane), z);
      sf[j] = z;
    }

    // online softmax: rowmax via ds_swizzle butterfly; exp2; rescale O
#pragma unroll
    for (int i = 0; i < 8; ++i) {
      float x = fmaxf(fmaxf(sf[0][i], sf[1][i]), fmaxf(sf[2][i], sf[3][i]));
      x = swz_max16(x);
      float mn = fmaxf(mrow[i], x);
      float alpha = exp2f(mrow[i] - mn);
      mrow[i] = mn;
      alphav[i] = alpha;
#pragma unroll
      for (int j = 0; j < 4; ++j) sf[j][i] = exp2f(sf[j][i] - mn);
#pragma unroll
      for (int j = 0; j < 4; ++j) o[j][i] *= alpha;
    }

    // C-layout f32 -> A-layout f16 via per-wave LDS staging
#pragma unroll
    for (int j = 0; j < 4; ++j)
#pragma unroll
      for (int i = 0; i < 8; ++i)
        myP[(i + 8 * (lane >> 4)) * 64 + j * 16 + (lane & 15)] = (_Float16)sf[j][i];
    __syncthreads();

    v16h pa0 = load_frag_a(myP, 64, lane);       // keys 0..31
    v16h pa1 = load_frag_a(myP + 32, 64, lane);  // keys 32..63

    // row sums: P @ ones -> every column (and thus every lane) holds its rows' sums
    v8f ls = {};
    ls = WMMA_F16(pa0, vones, ls);
    ls = WMMA_F16(pa1, vones, ls);
#pragma unroll
    for (int i = 0; i < 8; ++i) lrow[i] = lrow[i] * alphav[i] + ls[i];

#pragma unroll
    for (int j = 0; j < 4; ++j) {
      o[j] = WMMA_F16(pa0, load_frag_b(&sVt[buf][j * 16 * 64], 64, lane), o[j]);
      o[j] = WMMA_F16(pa1, load_frag_b(&sVt[buf][j * 16 * 64 + 32], 64, lane), o[j]);
    }
  }

  // epilogue: normalize (rcp + mul, f16 store follows), write [B*S, E]
  float linv[8];
#pragma unroll
  for (int i = 0; i < 8; ++i) linv[i] = __builtin_amdgcn_rcpf(lrow[i]);

  const int b = bh >> 3, h = bh & (NHEAD - 1);
  const int mh = 8 * (lane >> 4), nl = lane & 15;
#pragma unroll
  for (int j = 0; j < 4; ++j) {
#pragma unroll
    for (int i = 0; i < 8; ++i) {
      int s = q0 + i + mh;
      int d = j * 16 + nl;
      float v = o[j][i] * linv[i];
      Oout[((size_t)b * S_LEN + s) * EMB + h * HDIM + d] = (_Float16)v;
    }
  }
}

// ---------------------------------------------------------------------------
extern "C" void kernel_launch(void* const* d_in, const int* in_sizes, int n_in,
                              void* d_out, int out_size, void* d_ws, size_t ws_size,
                              hipStream_t stream) {
  const float* x  = (const float*)d_in[0];
  const float* Wq = (const float*)d_in[1];
  const float* Wk = (const float*)d_in[2];
  const float* Wv = (const float*)d_in[3];
  const float* Wo = (const float*)d_in[4];
  float* out = (float*)d_out;

  char* ws = (char*)d_ws;
  _Float16* xh  = (_Float16*)(ws);                      // 8 MB  [8192,512]
  _Float16* Wqh = (_Float16*)(ws + (size_t)(8 << 20));  // 4x 512KB weights
  _Float16* Wkh = Wqh + EMB * EMB;
  _Float16* Wvh = Wkh + EMB * EMB;
  _Float16* Woh = Wvh + EMB * EMB;
  _Float16* Qb  = (_Float16*)(ws + (size_t)(10 << 20)); // 8 MB [B*H,S,D]
  _Float16* Kb  = (_Float16*)(ws + (size_t)(18 << 20)); // 8 MB
  _Float16* Vb  = (_Float16*)(ws + (size_t)(26 << 20)); // 8 MB
  _Float16* Ah  = (_Float16*)(ws + (size_t)(34 << 20)); // 8 MB [8192,512]

  const int nx4 = MTOT * EMB / 4, nw4 = EMB * EMB / 4;
  cvt_f32_f16<<<(nx4 + 255) / 256, 256, 0, stream>>>(x, xh, nx4);
  cvt_f32_f16<<<(nw4 + 255) / 256, 256, 0, stream>>>(Wq, Wqh, nw4);
  cvt_f32_f16<<<(nw4 + 255) / 256, 256, 0, stream>>>(Wk, Wkh, nw4);
  cvt_f32_f16<<<(nw4 + 255) / 256, 256, 0, stream>>>(Wv, Wvh, nw4);
  cvt_f32_f16<<<(nw4 + 255) / 256, 256, 0, stream>>>(Wo, Woh, nw4);

  dim3 gproj(MTOT / 128, EMB / 128);
  // Q scaled by log2e/sqrt(D): softmax runs in log2 domain inside flash_attn.
  gemm_wmma<0><<<gproj, 256, 0, stream>>>(xh, Wqh, 0.125f * LOG2E, Qb, nullptr);
  gemm_wmma<0><<<gproj, 256, 0, stream>>>(xh, Wkh, 1.0f, Kb, nullptr);
  gemm_wmma<0><<<gproj, 256, 0, stream>>>(xh, Wvh, 1.0f, Vb, nullptr);

  flash_attn<<<dim3(S_LEN / 128, 2 * NHEAD), 256, 0, stream>>>(Qb, Kb, Vb, Ah);

  gemm_wmma<1><<<gproj, 256, 0, stream>>>(Ah, Woh, 1.0f, nullptr, out);
}
